// ProbSparseAttention_20444044329780
// MI455X (gfx1250) — matrix operand order repaired and checked by reading, not compiled, not run on hardware
//
#include <hip/hip_runtime.h>
#include <hip/hip_bf16.h>
#include <cstdint>
#include <cstddef>

// ---------------------------------------------------------------------------
// ProbSparse attention for MI455X (gfx1250, wave32, WMMA 16x16x32 bf16)
// B=2, L=4096, dm=128, H=8, D=16, sample_k=41
// ---------------------------------------------------------------------------

#define L_SEQ   4096
#define DM      128
#define NH      8
#define HD      16
#define NROWS   (2 * L_SEQ)        // 8192 total rows (B*L)
#define SKEYS   41                 // int(log(4096)*5)
#define SPAD    48                 // padded to 3 key tiles of 16
#define NEG_INF (-3.402823466e38f)

typedef __bf16 bf16_t;
typedef __attribute__((ext_vector_type(16))) bf16_t bf16x16;
typedef __attribute__((ext_vector_type(8)))  float  f32x8;

__device__ __forceinline__ f32x8 wmma_bf16(bf16x16 a, bf16x16 b, f32x8 c) {
    // D = A(16x32 bf16) * B(32x16 bf16) + C(16x16 f32)
    return __builtin_amdgcn_wmma_f32_16x16x32_bf16(
        /*neg_a=*/false, a, /*neg_b=*/false, b,
        /*c_mod=*/(short)0, c, /*reuse_a=*/false, /*reuse_b=*/false);
}

// A fragment, full K=32.  lane<16: K=0..7 in v[0..3], K=16..23 in v[4..7]
//                          lane>=16: caller passes base+8 -> K=8..15 / 24..31
__device__ __forceinline__ bf16x16 load_a32(const bf16_t* base) {
    bf16x16 v;
#pragma unroll
    for (int i = 0; i < 8; ++i) { v[i] = base[i]; v[8 + i] = base[16 + i]; }
    return v;
}

// A fragment with K padded to 32 (real K = 16): 8 real + 8 zero per lane
__device__ __forceinline__ bf16x16 load_a_pad16(const bf16_t* base) {
    bf16x16 v;
#pragma unroll
    for (int i = 0; i < 8; ++i) { v[i] = base[i]; v[8 + i] = (bf16_t)0.0f; }
    return v;
}

// B fragment: 16 contiguous bf16 (k-major run for this lane's column)
__device__ __forceinline__ bf16x16 load_b32(const bf16_t* base) {
    bf16x16 v;
#pragma unroll
    for (int i = 0; i < 16; ++i) v[i] = base[i];
    return v;
}

// B fragment with K padded to 32 (real K = 16): 8 real (or zeros for OOB keys) + 8 zero per lane
__device__ __forceinline__ bf16x16 load_brow16(const bf16_t* base, bool valid) {
    bf16x16 v;
#pragma unroll
    for (int i = 0; i < 8; ++i) { v[i] = valid ? base[i] : (bf16_t)0.0f; v[8 + i] = (bf16_t)0.0f; }
    return v;
}

// ---------------------------------------------------------------------------
// Kernel 1: fp32 -> bf16 elementwise
// ---------------------------------------------------------------------------
__global__ void k_convert_x(const float* __restrict__ x, bf16_t* __restrict__ xb, int n) {
    int j = blockIdx.x * blockDim.x + threadIdx.x;
    if (j < n) xb[j] = (bf16_t)x[j];
}

// ---------------------------------------------------------------------------
// Kernel 2: transpose+convert the four 128x128 weights: WT[n][k] = W[k][n]
// ---------------------------------------------------------------------------
__global__ void k_convert_wT(const float* __restrict__ Wq, const float* __restrict__ Wk,
                             const float* __restrict__ Wv, const float* __restrict__ Wo,
                             bf16_t* __restrict__ Tq, bf16_t* __restrict__ Tk,
                             bf16_t* __restrict__ Tv, bf16_t* __restrict__ To) {
    const float* W; bf16_t* T;
    switch (blockIdx.y) {
        case 0:  W = Wq; T = Tq; break;
        case 1:  W = Wk; T = Tk; break;
        case 2:  W = Wv; T = Tv; break;
        default: W = Wo; T = To; break;
    }
    int j = blockIdx.x * blockDim.x + threadIdx.x;   // j = k*128 + n
    if (j < DM * DM) {
        int k = j >> 7, n = j & 127;
        T[n * DM + k] = (bf16_t)W[j];
    }
}

// ---------------------------------------------------------------------------
// Kernel 3: fused QKV projection.  One wave computes one 16x16 output tile for
// all three projections (A fragment reused).  Q is pre-scaled by 1/sqrt(D).
// Outputs in [B,H,L,16] bf16 (row per token, head-dim contiguous).
// ---------------------------------------------------------------------------
__global__ void k_gemm_qkv(const bf16_t* __restrict__ xb,
                           const bf16_t* __restrict__ Tq, const bf16_t* __restrict__ Tk,
                           const bf16_t* __restrict__ Tv,
                           const float* __restrict__ bq, const float* __restrict__ bk,
                           const float* __restrict__ bv,
                           bf16_t* __restrict__ Q, bf16_t* __restrict__ K,
                           bf16_t* __restrict__ V) {
    const int lane = threadIdx.x & 31, wid = threadIdx.x >> 5;
    const int ln = lane & 15, grp = lane >> 4;
    const int tile = blockIdx.x * 8 + wid;          // 4096 tiles = 512 rowtiles x 8 heads
    const int rt = tile >> 3, ht = tile & 7;

    const bf16_t* wt[3] = {Tq, Tk, Tv};
    f32x8 acc[3] = {f32x8{}, f32x8{}, f32x8{}};

    const size_t arow = (size_t)(rt * 16 + ln) * DM;
#pragma unroll
    for (int t = 0; t < 4; ++t) {
        const bf16_t* abase = xb + arow + t * 32 + (grp ? 8 : 0);
        if (t < 3) __builtin_prefetch(abase + 32, 0, 1);
        bf16x16 a = load_a32(abase);
#pragma unroll
        for (int w = 0; w < 3; ++w) {
            const bf16_t* bbase = wt[w] + (size_t)(ht * 16 + ln) * DM + t * 32 + (grp ? 8 : 0);
            bf16x16 b = load_a32(bbase);
            acc[w] = wmma_bf16(a, b, acc[w]);
        }
    }

    const float biasv[3] = {bq[ht * 16 + ln], bk[ht * 16 + ln], bv[ht * 16 + ln]};
    bf16_t* outp[3] = {Q, K, V};
#pragma unroll
    for (int w = 0; w < 3; ++w) {
#pragma unroll
        for (int i = 0; i < 8; ++i) {
            int gr = rt * 16 + i + (grp ? 8 : 0);          // global token row
            int bb = gr >> 12, q = gr & (L_SEQ - 1);
            float v = acc[w][i] + biasv[w];
            if (w == 0) v *= 0.25f;                         // fold 1/sqrt(16) into Q
            outp[w][(((size_t)bb * NH + ht) * L_SEQ + q) * HD + ln] = (bf16_t)v;
        }
    }
}

// ---------------------------------------------------------------------------
// Kernel 4: per-query row max of scaled scores (streaming over all keys).
// grid = (qtile=256, bh=16); block = 256 (8 waves), each wave strides keytiles.
// ---------------------------------------------------------------------------
__global__ void k_rowmax(const bf16_t* __restrict__ Q, const bf16_t* __restrict__ K,
                         float* __restrict__ rowmax) {
    const int lane = threadIdx.x & 31, wid = threadIdx.x >> 5;
    const int ln = lane & 15, grp = lane >> 4;
    const int qt = blockIdx.x, bh = blockIdx.y;
    __shared__ float red[8][16];

    // Q fragment (K-dim 16 padded to 32), shared by all key tiles
    const bf16_t* qp = Q + ((size_t)bh * L_SEQ + qt * 16 + ln) * HD + (grp ? 8 : 0);
    bf16x16 a = load_a_pad16(qp);

    float mv[8];
#pragma unroll
    for (int i = 0; i < 8; ++i) mv[i] = NEG_INF;

    for (int kt = wid; kt < L_SEQ / 16; kt += 8) {
        const bf16_t* kp = K + ((size_t)bh * L_SEQ + kt * 16 + ln) * HD;
        __builtin_prefetch(K + ((size_t)bh * L_SEQ + (kt + 8) * 16 + ln) * HD, 0, 1);
        bf16x16 b = load_brow16(kp + (grp ? 8 : 0), true);      // elems 8-15 = K-pad zeros
        f32x8 zc = {};
        f32x8 s = wmma_bf16(a, b, zc);
#pragma unroll
        for (int i = 0; i < 8; ++i) mv[i] = fmaxf(mv[i], s[i]);
    }
    // reduce across the 16 lanes of each half (columns of the tile)
#pragma unroll
    for (int i = 0; i < 8; ++i) {
#pragma unroll
        for (int m = 1; m <= 8; m <<= 1) mv[i] = fmaxf(mv[i], __shfl_xor(mv[i], m, 32));
    }
    if (lane == 0)  { for (int i = 0; i < 8; ++i) red[wid][i]     = mv[i]; }
    if (lane == 16) { for (int i = 0; i < 8; ++i) red[wid][8 + i] = mv[i]; }
    __syncthreads();
    if (threadIdx.x < 16) {
        float m = NEG_INF;
        for (int w = 0; w < 8; ++w) m = fmaxf(m, red[w][threadIdx.x]);
        rowmax[(size_t)bh * L_SEQ + qt * 16 + threadIdx.x] = m;
    }
}

// ---------------------------------------------------------------------------
// Kernel 5: mean over queries of exp(score - rowmax[q]) per key (column sums).
// grid = (ktile=256, bh=16); each block owns 16 keys, streams all query tiles.
// ---------------------------------------------------------------------------
__global__ void k_colsum(const bf16_t* __restrict__ Q, const bf16_t* __restrict__ K,
                         const float* __restrict__ rowmax, float* __restrict__ colsum) {
    const int lane = threadIdx.x & 31, wid = threadIdx.x >> 5;
    const int ln = lane & 15, grp = lane >> 4;
    const int kt = blockIdx.x, bh = blockIdx.y;
    __shared__ float red[8][16];

    const bf16_t* kp = K + ((size_t)bh * L_SEQ + kt * 16 + ln) * HD;
    bf16x16 b = load_brow16(kp + (grp ? 8 : 0), true);

    float accL = 0.0f;
    for (int qt = wid; qt < L_SEQ / 16; qt += 8) {
        const bf16_t* qp = Q + ((size_t)bh * L_SEQ + qt * 16 + ln) * HD + (grp ? 8 : 0);
        __builtin_prefetch(Q + ((size_t)bh * L_SEQ + (qt + 8) * 16 + ln) * HD, 0, 1);
        bf16x16 a = load_a_pad16(qp);
        f32x8 zc = {};
        f32x8 s = wmma_bf16(a, b, zc);
        const float* rm = rowmax + (size_t)bh * L_SEQ + qt * 16 + (grp ? 8 : 0);
#pragma unroll
        for (int i = 0; i < 8; ++i) accL += __expf(s[i] - rm[i]);
    }
    accL += __shfl_xor(accL, 16, 32);               // rows 0-7 + rows 8-15
    if (lane < 16) red[wid][lane] = accL;
    __syncthreads();
    if (threadIdx.x < 16) {
        float s = 0.0f;
        for (int w = 0; w < 8; ++w) s += red[w][threadIdx.x];
        colsum[(size_t)bh * L_SEQ + kt * 16 + threadIdx.x] = s * (1.0f / L_SEQ);
    }
}

// ---------------------------------------------------------------------------
// Kernel 6: top-41 keys per (b,h) by iterative block-wide argmax.
// ---------------------------------------------------------------------------
__global__ void k_topk(const float* __restrict__ colsum, int* __restrict__ idx) {
    __shared__ float vals[L_SEQ];
    __shared__ float rv[256];
    __shared__ int   ri[256];
    const int bh = blockIdx.x, tid = threadIdx.x;
    for (int j = tid; j < L_SEQ; j += 256) vals[j] = colsum[(size_t)bh * L_SEQ + j];
    __syncthreads();
    for (int it = 0; it < SKEYS; ++it) {
        float best = NEG_INF; int bi = 0;
        for (int j = tid; j < L_SEQ; j += 256)
            if (vals[j] > best) { best = vals[j]; bi = j; }
        rv[tid] = best; ri[tid] = bi;
        __syncthreads();
        for (int s = 128; s > 0; s >>= 1) {
            if (tid < s && rv[tid + s] > rv[tid]) { rv[tid] = rv[tid + s]; ri[tid] = ri[tid + s]; }
            __syncthreads();
        }
        if (tid == 0) { idx[bh * SPAD + it] = ri[0]; vals[ri[0]] = NEG_INF; }
        __syncthreads();
    }
    if (tid < SPAD - SKEYS) idx[bh * SPAD + SKEYS + tid] = 0;   // pad (masked later)
}

// ---------------------------------------------------------------------------
// Kernel 7: sparse attention.  Per wave: one 16-query tile of one (b,h).
// scores (16x48, WMMA, gathered K rows) -> masked softmax -> probs & V^T
// staged in LDS as bf16 -> out = probs @ Vs via WMMA -> bf16 [B,L,dm].
// ---------------------------------------------------------------------------
__global__ void k_attn(const bf16_t* __restrict__ Q, const bf16_t* __restrict__ K,
                       const bf16_t* __restrict__ V, const int* __restrict__ idx,
                       bf16_t* __restrict__ x2) {
    const int lane = threadIdx.x & 31, wid = threadIdx.x >> 5;
    const int ln = lane & 15, grp = lane >> 4;
    const int bh = blockIdx.y, qt = blockIdx.x * 8 + wid;
    const int bb = bh >> 3, hh = bh & 7;

    __shared__ __align__(16) bf16_t smem[8 * 2048];
    bf16_t* VsT   = smem + wid * 2048;          // [16 d][64 s]
    bf16_t* probs = smem + wid * 2048 + 1024;   // [16 q][64 s]

    // gather V^T (zero-padded cols s>=41) and zero the probs buffer
    for (int s = lane; s < 64; s += 32) {
        int kid = (s < SKEYS) ? idx[bh * SPAD + s] : -1;
        const bf16_t* vp = V + ((size_t)bh * L_SEQ + (kid >= 0 ? kid : 0)) * HD;
#pragma unroll
        for (int d = 0; d < HD; ++d)
            VsT[d * 64 + s] = (kid >= 0) ? vp[d] : (bf16_t)0.0f;
    }
    for (int e = lane; e < 512; e += 32) ((unsigned int*)probs)[e] = 0u;
    __syncthreads();

    // Q fragment (already scaled by 1/sqrt(D)), K padded 16->32
    const bf16_t* qp = Q + ((size_t)bh * L_SEQ + qt * 16 + ln) * HD + (grp ? 8 : 0);
    bf16x16 a = load_a_pad16(qp);

    f32x8 sc[3];
#pragma unroll
    for (int nt = 0; nt < 3; ++nt) {
        int s = nt * 16 + ln;
        int kid = (s < SKEYS) ? idx[bh * SPAD + s] : -1;
        const bf16_t* kp = K + ((size_t)bh * L_SEQ + (kid >= 0 ? kid : 0)) * HD;
        bf16x16 b = load_brow16(kp + (grp ? 8 : 0), kid >= 0);
        f32x8 zc = {};
        sc[nt] = wmma_bf16(a, b, zc);
    }

    // masked softmax over 48 sampled keys (cols 41..47 invalid)
    const bool v2 = (ln < SKEYS - 32);              // nt=2 valid iff s=32+ln < 41
#pragma unroll
    for (int i = 0; i < 8; ++i) {
        float e0 = sc[0][i], e1 = sc[1][i];
        float e2 = v2 ? sc[2][i] : NEG_INF;
        float mx = fmaxf(e0, fmaxf(e1, e2));
#pragma unroll
        for (int m = 1; m <= 8; m <<= 1) mx = fmaxf(mx, __shfl_xor(mx, m, 32));
        e0 = __expf(e0 - mx); e1 = __expf(e1 - mx);
        e2 = v2 ? __expf(sc[2][i] - mx) : 0.0f;
        float sm = e0 + e1 + e2;
#pragma unroll
        for (int m = 1; m <= 8; m <<= 1) sm += __shfl_xor(sm, m, 32);
        float inv = 1.0f / sm;
        int r = i + (grp ? 8 : 0);
        probs[r * 64 + ln]      = (bf16_t)(e0 * inv);
        probs[r * 64 + 16 + ln] = (bf16_t)(e1 * inv);
        probs[r * 64 + 32 + ln] = (bf16_t)(e2 * inv);
    }
    __syncthreads();

    // out(16x16) = probs(16x64, zero-padded) @ VsT^T  via 2 bf16 WMMAs
    f32x8 co = {};
#pragma unroll
    for (int t = 0; t < 2; ++t) {
        bf16x16 af = load_a32(probs + ln * 64 + t * 32 + (grp ? 8 : 0));
        bf16x16 bf = load_a32(VsT + ln * 64 + t * 32 + (grp ? 8 : 0));
        co = wmma_bf16(af, bf, co);
    }
#pragma unroll
    for (int i = 0; i < 8; ++i) {
        int r = i + (grp ? 8 : 0);
        size_t grow = (size_t)bb * L_SEQ + qt * 16 + r;
        x2[grow * DM + hh * 16 + ln] = (bf16_t)co[i];
    }
}

// ---------------------------------------------------------------------------
// Kernel 8: output projection, fp32 result with bias -> d_out [B,L,dm]
// ---------------------------------------------------------------------------
__global__ void k_gemm_out(const bf16_t* __restrict__ x2, const bf16_t* __restrict__ To,
                           const float* __restrict__ bo, float* __restrict__ out) {
    const int lane = threadIdx.x & 31, wid = threadIdx.x >> 5;
    const int ln = lane & 15, grp = lane >> 4;
    const int tile = blockIdx.x * 8 + wid;
    const int rt = tile >> 3, ct = tile & 7;

    f32x8 acc = {};
    const size_t arow = (size_t)(rt * 16 + ln) * DM;
#pragma unroll
    for (int t = 0; t < 4; ++t) {
        const bf16_t* abase = x2 + arow + t * 32 + (grp ? 8 : 0);
        if (t < 3) __builtin_prefetch(abase + 32, 0, 1);
        bf16x16 a = load_a32(abase);
        bf16x16 b = load_a32(To + (size_t)(ct * 16 + ln) * DM + t * 32 + (grp ? 8 : 0));
        acc = wmma_bf16(a, b, acc);
    }
    const float bias = bo[ct * 16 + ln];
#pragma unroll
    for (int i = 0; i < 8; ++i) {
        int gr = rt * 16 + i + (grp ? 8 : 0);
        out[(size_t)gr * DM + ct * 16 + ln] = acc[i] + bias;
    }
}

// ---------------------------------------------------------------------------
// Launch
// ---------------------------------------------------------------------------
extern "C" void kernel_launch(void* const* d_in, const int* in_sizes, int n_in,
                              void* d_out, int out_size, void* d_ws, size_t ws_size,
                              hipStream_t stream) {
    const float* x  = (const float*)d_in[0];
    const float* Wq = (const float*)d_in[1];
    const float* bq = (const float*)d_in[2];
    const float* Wk = (const float*)d_in[3];
    const float* bk = (const float*)d_in[4];
    const float* Wv = (const float*)d_in[5];
    const float* bv = (const float*)d_in[6];
    const float* Wo = (const float*)d_in[7];
    const float* bo = (const float*)d_in[8];
    float* out = (float*)d_out;

    char* ws = (char*)d_ws;
    size_t off = 0;
    auto alloc = [&](size_t bytes) { char* p = ws + off; off = (off + bytes + 255) & ~(size_t)255; return p; };

    bf16_t* xb  = (bf16_t*)alloc((size_t)NROWS * DM * 2);
    bf16_t* Tq  = (bf16_t*)alloc((size_t)DM * DM * 2);
    bf16_t* Tk  = (bf16_t*)alloc((size_t)DM * DM * 2);
    bf16_t* Tv  = (bf16_t*)alloc((size_t)DM * DM * 2);
    bf16_t* To  = (bf16_t*)alloc((size_t)DM * DM * 2);
    bf16_t* Qb  = (bf16_t*)alloc((size_t)16 * L_SEQ * HD * 2);   // [B*H, L, 16]
    bf16_t* Kb  = (bf16_t*)alloc((size_t)16 * L_SEQ * HD * 2);
    bf16_t* Vb  = (bf16_t*)alloc((size_t)16 * L_SEQ * HD * 2);
    float*  rmx = (float*)alloc((size_t)16 * L_SEQ * 4);
    float*  csm = (float*)alloc((size_t)16 * L_SEQ * 4);
    int*    idx = (int*)alloc((size_t)16 * SPAD * 4);
    bf16_t* x2  = (bf16_t*)alloc((size_t)NROWS * DM * 2);
    (void)ws_size; (void)n_in; (void)in_sizes; (void)out_size;

    k_convert_x<<<(NROWS * DM + 255) / 256, 256, 0, stream>>>(x, xb, NROWS * DM);
    k_convert_wT<<<dim3((DM * DM + 255) / 256, 4), 256, 0, stream>>>(Wq, Wk, Wv, Wo, Tq, Tk, Tv, To);
    k_gemm_qkv<<<512, 256, 0, stream>>>(xb, Tq, Tk, Tv, bq, bk, bv, Qb, Kb, Vb);
    k_rowmax<<<dim3(L_SEQ / 16, 16), 256, 0, stream>>>(Qb, Kb, rmx);
    k_colsum<<<dim3(L_SEQ / 16, 16), 256, 0, stream>>>(Qb, Kb, rmx, csm);
    k_topk<<<16, 256, 0, stream>>>(csm, idx);
    k_attn<<<dim3(L_SEQ / 16 / 8, 16), 256, 0, stream>>>(Qb, Kb, Vb, idx, x2);
    k_gemm_out<<<512, 256, 0, stream>>>(x2, To, bo, out);
}
